// stagenet_selfcross_regcorr_91250875171466
// MI455X (gfx1250) — matrix operand order repaired and checked by reading, not compiled, not run on hardware
//
#include <hip/hip_runtime.h>
#include <hip/hip_bf16.h>
#include <math.h>

typedef __attribute__((ext_vector_type(16))) _Float16 v16h;
typedef __attribute__((ext_vector_type(8)))  float    v8f;

constexpr int V_ = 4, B_ = 2, C_ = 32, H_ = 128, W_ = 128, D_ = 32, G_ = 8, N_ = 4;
constexpr int HW_ = H_ * W_;

// ---------------- bilinear helper (zero padding, align_corners semantics) ----
struct BilinW {
    int o00, o01, o10, o11;       // pixel offsets (h*W+w units)
    float w00, w01, w10, w11;
};

__device__ __forceinline__ BilinW make_bilin(float x, float y) {
    BilinW bw;
    float x0f = floorf(x), y0f = floorf(y);
    float wx = x - x0f, wy = y - y0f;
    int x0 = (int)x0f, y0 = (int)y0f;
    int x1 = x0 + 1, y1 = y0 + 1;
    float vx0 = (x0 >= 0 && x0 < W_) ? 1.f : 0.f;
    float vx1 = (x1 >= 0 && x1 < W_) ? 1.f : 0.f;
    float vy0 = (y0 >= 0 && y0 < H_) ? 1.f : 0.f;
    float vy1 = (y1 >= 0 && y1 < H_) ? 1.f : 0.f;
    int x0c = min(max(x0, 0), W_ - 1), x1c = min(max(x1, 0), W_ - 1);
    int y0c = min(max(y0, 0), H_ - 1), y1c = min(max(y1, 0), H_ - 1);
    bw.o00 = y0c * W_ + x0c; bw.o01 = y0c * W_ + x1c;
    bw.o10 = y1c * W_ + x0c; bw.o11 = y1c * W_ + x1c;
    bw.w00 = vx0 * vy0 * (1.f - wx) * (1.f - wy);
    bw.w01 = vx1 * vy0 * wx * (1.f - wy);
    bw.w10 = vx0 * vy1 * (1.f - wx) * wy;
    bw.w11 = vx1 * vy1 * wx * wy;
    return bw;
}

// sample with an element stride (stride=1: plain HxW image; stride=D: [p][d] layout)
__device__ __forceinline__ float bilin_sample_s(const float* __restrict__ ch,
                                                const BilinW& b, int s) {
    return b.w00 * ch[(size_t)b.o00 * s] + b.w01 * ch[(size_t)b.o01 * s] +
           b.w10 * ch[(size_t)b.o10 * s] + b.w11 * ch[(size_t)b.o11 * s];
}

// ---------------- projection setup: rot/trans per (v,b) ---------------------
// proj = src_proj_new @ inv(ref_proj_new); both are affine [[M,t],[0,1]].
__global__ void proj_kernel(const float* __restrict__ projs, float* __restrict__ ws_proj) {
    int t = threadIdx.x;
    if (t >= V_ * B_) return;
    int v = t / B_, b = t % B_;
    auto M4 = [&](int view, int m, int r, int c) -> float {
        return projs[(((size_t)b * (V_ + 1) + view) * 2 + m) * 16 + r * 4 + c];
    };
    float Mr[9], tr[3];
    for (int r = 0; r < 3; r++) {
        for (int j = 0; j < 3; j++) {
            float s = 0.f;
            for (int k = 0; k < 3; k++) s += M4(0, 1, r, k) * M4(0, 0, k, j);
            Mr[r * 3 + j] = s;
        }
        float s = 0.f;
        for (int k = 0; k < 3; k++) s += M4(0, 1, r, k) * M4(0, 0, k, 3);
        tr[r] = s;
    }
    float a = Mr[0], bb = Mr[1], c = Mr[2], d = Mr[3], e = Mr[4], f = Mr[5],
          g = Mr[6], h = Mr[7], i = Mr[8];
    float det = a * (e * i - f * h) - bb * (d * i - f * g) + c * (d * h - e * g);
    float id = 1.0f / det;
    float Mi[9] = { (e * i - f * h) * id, (c * h - bb * i) * id, (bb * f - c * e) * id,
                    (f * g - d * i) * id, (a * i - c * g) * id,  (c * d - a * f) * id,
                    (d * h - e * g) * id, (bb * g - a * h) * id, (a * e - bb * d) * id };
    float ti[3];
    for (int r = 0; r < 3; r++) {
        float s = 0.f;
        for (int k = 0; k < 3; k++) s += Mi[r * 3 + k] * tr[k];
        ti[r] = -s;
    }
    float As[9], bs[3];
    for (int r = 0; r < 3; r++) {
        for (int j = 0; j < 3; j++) {
            float s = 0.f;
            for (int k = 0; k < 3; k++) s += M4(v + 1, 1, r, k) * M4(v + 1, 0, k, j);
            As[r * 3 + j] = s;
        }
        float s = 0.f;
        for (int k = 0; k < 3; k++) s += M4(v + 1, 1, r, k) * M4(v + 1, 0, k, 3);
        bs[r] = s;
    }
    float* out = ws_proj + (size_t)(v * B_ + b) * 12;
    for (int r = 0; r < 3; r++)
        for (int j = 0; j < 3; j++) {
            float s = 0.f;
            for (int k = 0; k < 3; k++) s += As[r * 3 + k] * Mi[k * 3 + j];
            out[r * 3 + j] = s;
        }
    for (int r = 0; r < 3; r++) {
        float s = 0.f;
        for (int k = 0; k < 3; k++) s += As[r * 3 + k] * ti[k];
        out[9 + r] = s + bs[r];
    }
}

// ---------------- accumulator init ------------------------------------------
__global__ void init_kernel(float* __restrict__ acc_wl, float* __restrict__ acc_sum) {
    int idx = blockIdx.x * blockDim.x + threadIdx.x;
    acc_wl[idx] = 0.f;
    acc_sum[idx] = 1e-8f;
}

// ---------------- kernel A: homography warp + group correlation -------------
// one thread per (b,p,d) with d FASTEST (lane == depth):
//   * projection math + ref loads are wave-uniform
//   * cor0 stores [b][g][p][d] are fully coalesced across the wave
__global__ void warp_corr_kernel(const float* __restrict__ ref_fea,
                                 const float* __restrict__ src_fea,
                                 const float* __restrict__ depth_hypo,
                                 const float* __restrict__ ws_proj,
                                 float* __restrict__ cor0, int v) {
    int idx = blockIdx.x * blockDim.x + threadIdx.x;   // < B*HW*D
    int d = idx & (D_ - 1);
    int p = (idx >> 5) & (HW_ - 1);
    int b = idx >> 19;
    int h = p >> 7, w = p & (W_ - 1);
    const float* pr = ws_proj + (size_t)(v * B_ + b) * 12;
    float depth = depth_hypo[(size_t)(b * D_ + d) * HW_ + p];
    float fx = (float)w, fy = (float)h;
    float rx = pr[0] * fx + pr[1] * fy + pr[2];
    float ry = pr[3] * fx + pr[4] * fy + pr[5];
    float rz = pr[6] * fx + pr[7] * fy + pr[8];
    float pz = rz * depth + pr[11];
    if (pz == 0.f) pz = 1e-9f;
    // (gx+1)*0.5*(Ws-1) round-trips to plain pixel coords with align_corners
    float x = (rx * depth + pr[9]) / pz;
    float y = (ry * depth + pr[10]) / pz;
    BilinW bw = make_bilin(x, y);
    const float* srcb = src_fea + (size_t)(v * B_ + b) * C_ * HW_;
    const float* refb = ref_fea + (size_t)(v * B_ + b) * C_ * HW_ + p;
    float acc[G_];
#pragma unroll
    for (int g = 0; g < G_; g++) acc[g] = 0.f;
#pragma unroll
    for (int c = 0; c < C_; c++) {
        float s = bilin_sample_s(srcb + (size_t)c * HW_, bw, 1);
        acc[c >> 2] += s * refb[(size_t)c * HW_];
    }
    // cor0 layout: [b][g][p][d]  (d contiguous)
    size_t ob = ((size_t)b * G_ * HW_ + p) * D_ + d;
#pragma unroll
    for (int g = 0; g < G_; g++) cor0[ob + (size_t)g * HW_ * D_] = acc[g] * 0.25f;
}

// ---------------- kernel B: enhancement GEMM (WMMA) + softmax accumulate ----
// one wave per pixel (b,h,w); lane = depth d (D == wave32).
// cor0 is [b][g][p][d] so every gather is coalesced over lanes.
__global__ void enhance_kernel(const float* __restrict__ cor0,
                               const float* __restrict__ grids,
                               const float* __restrict__ w_enh,
                               const float* __restrict__ w_reg,
                               float* __restrict__ acc_wl,
                               float* __restrict__ acc_sum,
                               float* __restrict__ out_corw, int v) {
    __shared__ unsigned s_cfs[8][32][16];   // [wave][depth row][K word] 2xf16 packed
    __shared__ float    s_out[8][32][8];    // [wave][depth row][g]
    int lane = threadIdx.x & 31;
    int wv = threadIdx.x >> 5;
    int pix = blockIdx.x * 8 + wv;          // < B*HW
    int b = pix >> 14;
    int p = pix & (HW_ - 1);
    int d = lane;
    const float* c0b = cor0 + (size_t)b * G_ * HW_ * D_;
    float base[G_];
#pragma unroll
    for (int g = 0; g < G_; g++) base[g] = c0b[((size_t)g * HW_ + p) * D_ + d];

    // cf_s[k], k = g'*N + n : bilinear re-sample of cor_feat0 at grid points
    float cfs[G_ * N_];
#pragma unroll
    for (int n = 0; n < N_; n++) {
        const float* gp = grids + ((((size_t)v * B_ + b) * N_ + n) * HW_ + p) * 2;
        float x = (gp[0] + 1.f) * 0.5f * (W_ - 1);
        float y = (gp[1] + 1.f) * 0.5f * (H_ - 1);
        BilinW bw = make_bilin(x, y);
#pragma unroll
        for (int gg = 0; gg < G_; gg++)
            cfs[gg * N_ + n] = bilin_sample_s(c0b + (size_t)gg * HW_ * D_ + d, bw, D_);
    }
    // stage as packed f16 pairs (word k = {cfs[2k] lo, cfs[2k+1] hi})
    union H2 { _Float16 h[2]; unsigned u; };
#pragma unroll
    for (int k = 0; k < 16; k++) {
        H2 t;
        t.h[0] = (_Float16)cfs[2 * k];
        t.h[1] = (_Float16)cfs[2 * k + 1];
        s_cfs[wv][lane][k] = t.u;
    }
    __syncthreads();

    // A fragments (ISA 16-bit A 16x32): lane<16 holds K{0..7,16..23},
    // lane>=16 holds K{8..15,24..31}; row M = lane&15 (+16 for 2nd tile).
    int laneHi = lane >> 4;
    int row = lane & 15;
    union U { v16h v; _Float16 e[16]; unsigned u32[8]; };
    U a0, a1, bf;
#pragma unroll
    for (int k2 = 0; k2 < 4; k2++) {
        a0.u32[k2]     = s_cfs[wv][row][laneHi * 4 + k2];           // K = laneHi*8 + 2k2(+1)
        a0.u32[4 + k2] = s_cfs[wv][row][8 + laneHi * 4 + k2];       // K = 16 + laneHi*8 + ...
        a1.u32[k2]     = s_cfs[wv][row + 16][laneHi * 4 + k2];
        a1.u32[4 + k2] = s_cfs[wv][row + 16][8 + laneHi * 4 + k2];
    }
    // B fragment: B[k,g] = w_enh[g,k]; lanes 0-15 carry K{0..7,16..23}, 16-31 K{8..15,24..31}.
    // Unconditional load from clamped row + 0/1 mask (no EXEC divergence).
    int col = row;
    int crow = col & (G_ - 1);
    float msk = (col < G_) ? 1.f : 0.f;
    int kbase = laneHi * 8;
    const float* wrow = w_enh + crow * (G_ * N_) + kbase;
#pragma unroll
    for (int e = 0; e < 16; e++)
        bf.e[e] = (_Float16)(msk * wrow[e < 8 ? e : e + 8]);

    v8f cz = {};
    v8f d0 = __builtin_amdgcn_wmma_f32_16x16x32_f16(false, a0.v, false, bf.v,
                                                    (short)0, cz, false, false);
    v8f d1 = __builtin_amdgcn_wmma_f32_16x16x32_f16(false, a1.v, false, bf.v,
                                                    (short)0, cz, false, false);
    // D layout: lane = col N, VGPR r = row M (lanes 0-15: M=r; 16-31: M=8+r)
    if (col < G_) {
#pragma unroll
        for (int r = 0; r < 8; r++) {
            int r0 = laneHi ? (8 + r) : r;
            s_out[wv][r0][col]      = d0[r];
            s_out[wv][16 + r0][col] = d1[r];
        }
    }
    __syncthreads();

    float ssum = 0.f, wl = 0.f;
#pragma unroll
    for (int g = 0; g < G_; g++) {
        float f = base[g] + s_out[wv][d][g];
        ssum += f;
        wl += w_reg[g] * f;
    }
    float logit = ssum * 0.5f;  // / ATTN_TEMP
    float m = logit;
#pragma unroll
    for (int off = 16; off >= 1; off >>= 1) m = fmaxf(m, __shfl_xor(m, off));
    float ex = __expf(logit - m);
    float se = ex;
#pragma unroll
    for (int off = 16; off >= 1; off >>= 1) se += __shfl_xor(se, off);
    float cw = ex / se * 0.17677669529663687f;  // 1/sqrt(C)
    // acc layout: [b][p][d]  (d contiguous -> coalesced RMW)
    size_t aidx = ((size_t)b * HW_ + p) * D_ + d;
    acc_sum[aidx] += cw;
    acc_wl[aidx]  += cw * wl;   // accumulate w_reg-projected logit (linearized)
    float cwm = cw;
#pragma unroll
    for (int off = 16; off >= 1; off >>= 1) cwm = fmaxf(cwm, __shfl_xor(cwm, off));
    if (lane == 0) out_corw[((size_t)v * B_ + b) * HW_ + p] = cwm;
}

// ---------------- kernel C: final softmax / argmax / outputs ----------------
__global__ void final_kernel(const float* __restrict__ acc_wl,
                             const float* __restrict__ acc_sum,
                             const float* __restrict__ depth_hypo,
                             float* __restrict__ out_depth,
                             float* __restrict__ out_conf,
                             float* __restrict__ out_attn) {
    int lane = threadIdx.x & 31;
    int wv = threadIdx.x >> 5;
    int pix = blockIdx.x * 8 + wv;
    int b = pix >> 14;
    int p = pix & (HW_ - 1);
    int d = lane;
    size_t aidx = ((size_t)b * HW_ + p) * D_ + d;    // coalesced reads
    float logit = acc_wl[aidx] / acc_sum[aidx];
    float m = logit;
#pragma unroll
    for (int off = 16; off >= 1; off >>= 1) m = fmaxf(m, __shfl_xor(m, off));
    float ex = __expf(logit - m);
    float se = ex;
#pragma unroll
    for (int off = 16; off >= 1; off >>= 1) se += __shfl_xor(se, off);
    float aw = ex / se;
    out_attn[(size_t)(b * D_ + d) * HW_ + p] = aw;   // output layout (B,D,H,W)
    // argmax with first-index tie-break
    float bv = aw;
    int bi = d;
#pragma unroll
    for (int off = 16; off >= 1; off >>= 1) {
        float ov = __shfl_xor(bv, off);
        int oi = __shfl_xor(bi, off);
        if (ov > bv || (ov == bv && oi < bi)) { bv = ov; bi = oi; }
    }
    if (lane == 0) {
        out_conf[(size_t)b * HW_ + p] = bv;
        out_depth[(size_t)b * HW_ + p] = depth_hypo[(size_t)(b * D_ + bi) * HW_ + p];
    }
}

// ---------------- launch -----------------------------------------------------
extern "C" void kernel_launch(void* const* d_in, const int* in_sizes, int n_in,
                              void* d_out, int out_size, void* d_ws, size_t ws_size,
                              hipStream_t stream) {
    (void)in_sizes; (void)n_in; (void)out_size; (void)ws_size;
    const float* ref        = (const float*)d_in[0];
    const float* src        = (const float*)d_in[1];
    const float* projs      = (const float*)d_in[2];
    const float* depth_hypo = (const float*)d_in[3];
    const float* grids      = (const float*)d_in[4];
    const float* w_reg      = (const float*)d_in[5];
    const float* w_enh      = (const float*)d_in[6];

    float* ws = (float*)d_ws;
    float* ws_proj = ws;                                      // 96 floats
    float* cor0 = ws + 128;                                   // [b][g][p][d] (32 MB)
    size_t cor0_sz = (size_t)B_ * G_ * HW_ * D_;
    float* acc_wl = cor0 + cor0_sz;                           // [b][p][d] (4 MB)
    float* acc_sum = acc_wl + (size_t)B_ * HW_ * D_;          // [b][p][d] (4 MB)

    float* out = (float*)d_out;
    float* out_depth = out;
    float* out_conf  = out + (size_t)B_ * HW_;
    float* out_attn  = out + (size_t)2 * B_ * HW_;
    float* out_corw  = out_attn + (size_t)B_ * D_ * HW_;

    proj_kernel<<<1, 32, 0, stream>>>(projs, ws_proj);
    init_kernel<<<(B_ * HW_ * D_) / 256, 256, 0, stream>>>(acc_wl, acc_sum);
    for (int v = 0; v < V_; ++v) {
        warp_corr_kernel<<<(B_ * HW_ * D_) / 256, 256, 0, stream>>>(
            ref, src, depth_hypo, ws_proj, cor0, v);
        enhance_kernel<<<(B_ * HW_) / 8, 256, 0, stream>>>(
            cor0, grids, w_enh, w_reg, acc_wl, acc_sum, out_corw, v);
    }
    final_kernel<<<(B_ * HW_) / 8, 256, 0, stream>>>(
        acc_wl, acc_sum, depth_hypo, out_depth, out_conf, out_attn);
}
